// TernaryPackedLinear_75067438400007
// MI455X (gfx1250) — compile-verified, matrix operand order checked by reading, not run on hardware
//
#include <hip/hip_runtime.h>

typedef __attribute__((ext_vector_type(16))) _Float16 v16h;
typedef __attribute__((ext_vector_type(8)))  float    v8f;

#define KK     4096
#define NN     16384
#define NWORDS 512
#define KSTEPS (KK / 32)
#define SPLITK 4

union HVec {
    v16h h;
    int4 q[2];
    int  i[8];
};

// Decode one k-row's bits for BOTH n-tiles at once.
// t = nz>>lm has tile0's bit at pos 0, tile1's bit at pos 16.
// Result: [15:0] = f16 weight for (row, n0+lm), [31:16] = f16 weight for (row, n0+16+lm).
static __device__ __forceinline__ unsigned decode_pair(int nzr, int sgr, int lm) {
    unsigned t = ((unsigned)nzr) >> lm;
    unsigned g = t & (((unsigned)sgr) >> lm);
    return ((t & 0x10001u) * 0x3C00u) | ((g & 0x10001u) << 15); // carry-free dual scale
}

template<bool FUSED>
__global__ __launch_bounds__(256) void ternary_gemm_wmma(
    const _Float16* __restrict__ x,
    const int*      __restrict__ Bnz,
    const int*      __restrict__ Bsg,
    const float*    __restrict__ bias,
    _Float16*       __restrict__ out,
    float*          __restrict__ ws,
    int ks_per_part)
{
    const int lane  = threadIdx.x & 31;
    const int wave  = threadIdx.x >> 5;
    const int chunk = blockIdx.x * 8 + wave;   // one 32-column slab of N per wave
    const int part  = blockIdx.y;              // K partition
    const int n0    = chunk * 32;
    const int w     = chunk;                   // packed word column (n0/32)
    const int half  = lane >> 4;
    const int lm    = lane & 15;

    const _Float16* xr0 = x + lm * KK;          // M tile 0: m = lm
    const _Float16* xr1 = x + (lm + 16) * KK;   // M tile 1: m = 16 + lm

    v8f c00 = {}, c01 = {}, c10 = {}, c11 = {};

    const int ks_begin = part * ks_per_part;
    const int ks_end   = ks_begin + ks_per_part;

    #pragma unroll 2
    for (int ks = ks_begin; ks < ks_end; ++ks) {
        const int kbase = ks * 32;
        const int kA    = kbase + half * 8;     // this lane's K group base

        // --- A fragments (documented 16-bit A layout): K kA..kA+7 then kA+16..kA+23
        HVec a0, a1;
        a0.q[0] = *(const int4*)(xr0 + kA);
        a0.q[1] = *(const int4*)(xr0 + kA + 16);
        a1.q[0] = *(const int4*)(xr1 + kA);
        a1.q[1] = *(const int4*)(xr1 + kA + 16);

        // --- lane i fetches packed words for k-row kbase+i, word column w
        const int nzw = Bnz[(kbase + lane) * NWORDS + w];
        const int sgw = Bsg[(kbase + lane) * NWORDS + w];

        // --- build both B fragments; one perm per fragment dword
        HVec b0, b1;
        #pragma unroll
        for (int jj = 0; jj < 4; ++jj) {
            const int s0 = (half * 8 + 2 * jj) << 2;        // lane holding row kA+2jj
            const int s1 = s0 + 4;                          // row kA+2jj+1
            const int s2 = (16 + half * 8 + 2 * jj) << 2;   // row kA+16+2jj
            const int s3 = s2 + 4;                          // row kA+16+2jj+1
            const unsigned R0 = decode_pair(__builtin_amdgcn_ds_bpermute(s0, nzw),
                                            __builtin_amdgcn_ds_bpermute(s0, sgw), lm);
            const unsigned R1 = decode_pair(__builtin_amdgcn_ds_bpermute(s1, nzw),
                                            __builtin_amdgcn_ds_bpermute(s1, sgw), lm);
            const unsigned R2 = decode_pair(__builtin_amdgcn_ds_bpermute(s2, nzw),
                                            __builtin_amdgcn_ds_bpermute(s2, sgw), lm);
            const unsigned R3 = decode_pair(__builtin_amdgcn_ds_bpermute(s3, nzw),
                                            __builtin_amdgcn_ds_bpermute(s3, sgw), lm);
            // low halves of (R0,R1) -> n-tile0 dword; high halves -> n-tile1 dword
            b0.i[jj]     = __builtin_amdgcn_perm(R1, R0, 0x05040100);
            b1.i[jj]     = __builtin_amdgcn_perm(R1, R0, 0x07060302);
            b0.i[4 + jj] = __builtin_amdgcn_perm(R3, R2, 0x05040100);
            b1.i[4 + jj] = __builtin_amdgcn_perm(R3, R2, 0x07060302);
        }

        // --- 2x2 tile of 16x16x32 WMMAs, f32 accumulation (matches reference numerics)
        c00 = __builtin_amdgcn_wmma_f32_16x16x32_f16(false, a0.h, false, b0.h, (short)0, c00, false, false);
        c01 = __builtin_amdgcn_wmma_f32_16x16x32_f16(false, a0.h, false, b1.h, (short)0, c01, false, false);
        c10 = __builtin_amdgcn_wmma_f32_16x16x32_f16(false, a1.h, false, b0.h, (short)0, c10, false, false);
        c11 = __builtin_amdgcn_wmma_f32_16x16x32_f16(false, a1.h, false, b1.h, (short)0, c11, false, false);
    }

    // C layout: lane L, vgpr j -> M = j + 8*(L>=16), N = L&15
    const int nA = n0 + lm;
    const int nB = nA + 16;

    if (FUSED) {
        const float bA = bias[nA];
        const float bB = bias[nB];
        #pragma unroll
        for (int j = 0; j < 8; ++j) {
            const int m0 = j + half * 8;
            const int m1 = m0 + 16;
            float f;
            f = c00[j] + bA; f = fmaxf(f, 0.0f); out[m0 * NN + nA] = (_Float16)f;
            f = c01[j] + bB; f = fmaxf(f, 0.0f); out[m0 * NN + nB] = (_Float16)f;
            f = c10[j] + bA; f = fmaxf(f, 0.0f); out[m1 * NN + nA] = (_Float16)f;
            f = c11[j] + bB; f = fmaxf(f, 0.0f); out[m1 * NN + nB] = (_Float16)f;
        }
    } else {
        // partials: ws[part][m][n]; lanes hold consecutive n -> stores coalesce
        float* wsp = ws + (size_t)part * 32 * NN;
        #pragma unroll
        for (int j = 0; j < 8; ++j) {
            const int m0 = j + half * 8;
            const int m1 = m0 + 16;
            wsp[m0 * NN + nA] = c00[j];
            wsp[m0 * NN + nB] = c01[j];
            wsp[m1 * NN + nA] = c10[j];
            wsp[m1 * NN + nB] = c11[j];
        }
    }
}

__global__ __launch_bounds__(256) void reduce_bias_relu(
    const float* __restrict__ ws,
    const float* __restrict__ bias,
    _Float16*    __restrict__ out)
{
    const int id = blockIdx.x * 256 + threadIdx.x;  // over 32*NN outputs
    const int n  = id & (NN - 1);
    const int m  = id >> 14;                        // NN == 1<<14
    float s = ws[(0 * 32 + m) * NN + n]
            + ws[(1 * 32 + m) * NN + n]
            + ws[(2 * 32 + m) * NN + n]
            + ws[(3 * 32 + m) * NN + n];
    s += bias[n];
    s = fmaxf(s, 0.0f);
    out[id] = (_Float16)s;
}

extern "C" void kernel_launch(void* const* d_in, const int* in_sizes, int n_in,
                              void* d_out, int out_size, void* d_ws, size_t ws_size,
                              hipStream_t stream) {
    (void)in_sizes; (void)n_in; (void)out_size;
    const _Float16* x    = (const _Float16*)d_in[0];
    const int*      Bnz  = (const int*)d_in[1];
    const int*      Bsg  = (const int*)d_in[2];
    const float*    bias = (const float*)d_in[3];
    _Float16*       out  = (_Float16*)d_out;
    float*          ws   = (float*)d_ws;

    const size_t ws_needed = (size_t)SPLITK * 32 * NN * sizeof(float);  // 8 MB
    dim3 block(256);

    if (ws_size >= ws_needed) {
        // split-K: 64 x 4 blocks = 2048 waves, then deterministic reduction
        dim3 grid(NN / 32 / 8, SPLITK);
        ternary_gemm_wmma<false><<<grid, block, 0, stream>>>(x, Bnz, Bsg, bias, out, ws,
                                                             KSTEPS / SPLITK);
        dim3 rgrid((32 * NN) / 256);
        reduce_bias_relu<<<rgrid, block, 0, stream>>>(ws, bias, out);
    } else {
        // fallback: fully fused single pass
        dim3 grid(NN / 32 / 8, 1);
        ternary_gemm_wmma<true><<<grid, block, 0, stream>>>(x, Bnz, Bsg, bias, out, ws,
                                                            KSTEPS);
    }
}
